// MultiHeadAttention_8693013807554
// MI455X (gfx1250) — matrix operand order repaired and checked by reading, not compiled, not run on hardware
//
#include <hip/hip_runtime.h>
#include <hip/hip_bf16.h>

// ---------------------------------------------------------------------------
// MultiHeadAttention with LoRA (Q,V) for MI455X / gfx1250 (wave32, WMMA).
//   B=2, S=2048, E=1024, H=16, D=64, R=8
// Pipeline:
//   1) convert q/k/v fp32 -> bf16; transpose+convert weights to bf16 [N][K]
//      (LoRA folded into Wq/Wv during the transpose)
//   2) projections: double-buffered async-LDS bf16 WMMA GEMM (+bias)
//      (V projection stores transposed per head: Vt[b][h][d][s])
//   3) flash attention, TK=64 key tiles, async double-buffered K/V staging,
//      fp32 online softmax, scale folded into Q fragments
//   4) output projection -> fp32 d_out
// ---------------------------------------------------------------------------

typedef __attribute__((ext_vector_type(16))) __bf16 v16bf;
typedef __attribute__((ext_vector_type(8)))  __bf16 v8bf;
typedef __attribute__((ext_vector_type(4)))  __bf16 v4bf;
typedef __attribute__((ext_vector_type(8)))  float  v8f;
typedef __attribute__((ext_vector_type(4)))  int    v4i;

constexpr int Bc = 2;
constexpr int Sc = 2048;
constexpr int Ec = 1024;
constexpr int Hc = 16;
constexpr int Dc = 64;
constexpr int Rc = 8;
constexpr int Mc = Bc * Sc;   // 4096 token rows

static __device__ __forceinline__ v8f wmma_bf16(v16bf a, v16bf b, v8f c) {
  return __builtin_amdgcn_wmma_f32_16x16x32_bf16(
      /*neg_a=*/false, a, /*neg_b=*/false, b,
      /*c_mod=*/(short)0, c, /*reuse_a=*/false, /*reuse_b=*/false);
}

// ---- gfx1250 async global->LDS copy (16B per lane), guarded ---------------
#if defined(__AMDGCN__) && __has_builtin(__builtin_amdgcn_global_load_async_to_lds_b128)
#define ASYNC_LDS_OK 1
#else
#define ASYNC_LDS_OK 0
#endif

#if ASYNC_LDS_OK
typedef __attribute__((address_space(1))) v4i gv4i;   // global int4
typedef __attribute__((address_space(3))) v4i lv4i;   // LDS int4
#endif

static __device__ __forceinline__ void async_cp16(const __bf16* g, __bf16* l) {
#if ASYNC_LDS_OK
  __builtin_amdgcn_global_load_async_to_lds_b128((gv4i*)(g), (lv4i*)(l), 0, 0);
#else
  *(v8bf*)l = *(const v8bf*)g;
#endif
}

static __device__ __forceinline__ void wait_async0() {
#if ASYNC_LDS_OK
#if __has_builtin(__builtin_amdgcn_s_wait_asynccnt)
  __builtin_amdgcn_s_wait_asynccnt(0);
#else
  asm volatile("s_wait_asynccnt 0x0" ::: "memory");
#endif
#endif
}

// ---------------------------------------------------------------------------
// fp32 -> bf16 streaming convert (n % 4 == 0)
// ---------------------------------------------------------------------------
__global__ __launch_bounds__(256)
void cvt_bf16_kernel(const float* __restrict__ x, __bf16* __restrict__ y, int n) {
  int i = (blockIdx.x * 256 + threadIdx.x) * 4;
  if (i >= n) return;
  float4 f = *(const float4*)(x + i);
  v4bf o;
  o[0] = (__bf16)f.x; o[1] = (__bf16)f.y; o[2] = (__bf16)f.z; o[3] = (__bf16)f.w;
  *(v4bf*)(y + i) = o;
}

// ---------------------------------------------------------------------------
// Weight transpose + convert: WT[n][k] = (bf16) W[k][n]   (E x E)
// ---------------------------------------------------------------------------
__global__ __launch_bounds__(256)
void wtrans_kernel(const float* __restrict__ W, __bf16* __restrict__ WT) {
  __shared__ float tile[32][33];
  const int bx = blockIdx.x * 32;   // n block
  const int by = blockIdx.y * 32;   // k block
  const int tx = threadIdx.x & 31;
  const int ty = threadIdx.x >> 5;  // 0..7
  for (int i = ty; i < 32; i += 8)
    tile[i][tx] = W[(size_t)(by + i) * Ec + bx + tx];
  __syncthreads();
  for (int i = ty; i < 32; i += 8)
    WT[(size_t)(bx + i) * Ec + by + tx] = (__bf16)tile[tx][i];
}

// ---------------------------------------------------------------------------
// LoRA fold + transpose: WT[n][k] = (bf16)(W[k][n] + sum_r A[k][r]*Bm[r][n])
// ---------------------------------------------------------------------------
__global__ __launch_bounds__(256)
void lora_fold_t_kernel(const float* __restrict__ W, const float* __restrict__ A,
                        const float* __restrict__ Bm, __bf16* __restrict__ WT) {
  int idx = blockIdx.x * 256 + threadIdx.x;   // idx = k*E + n
  if (idx >= Ec * Ec) return;
  int kk = idx / Ec;
  int nn = idx % Ec;
  float acc = W[idx];
#pragma unroll
  for (int rr = 0; rr < Rc; ++rr)
    acc += A[kk * Rc + rr] * Bm[rr * Ec + nn];
  WT[(size_t)nn * Ec + kk] = (__bf16)acc;
}

// ---------------------------------------------------------------------------
// Double-buffered WMMA GEMM:  C[M,N] = A[M,K] @ WT[N,K]^T + bias[N]
//   A, WT bf16 row-major. 128x128 tile, BK=64, 256 thr = 8 waves (4x2),
//   wave patch 32x64 = 2x4 WMMA tiles. All staging via async-to-LDS b128.
//   TRANSOUT: scatter output as Vt[b][h][d][s] (for the V projection).
// ---------------------------------------------------------------------------
constexpr int GBM = 128, GBN = 128, GBK = 64;

template <typename TOUT, bool TRANSOUT>
__global__ __launch_bounds__(256)
void gemm_bias_kernel(const __bf16* __restrict__ A, const __bf16* __restrict__ WT,
                      const float* __restrict__ bias, TOUT* __restrict__ C,
                      int M, int N, int K) {
  __shared__ __bf16 sA[2][GBM][GBK + 8];   // 144B row stride (16B aligned)
  __shared__ __bf16 sB[2][GBN][GBK + 8];

  const int tid  = threadIdx.x;
  const int wave = tid >> 5;
  const int lane = tid & 31;
  const int half = lane >> 4;
  const int r    = lane & 15;
  const int wm   = wave >> 1;    // 0..3
  const int wn   = wave & 1;     // 0..1

  const int m0 = blockIdx.y * GBM;
  const int n0 = blockIdx.x * GBN;

  auto stage = [&](int buf, int k0) {
    // A tile: 128 rows x 64 k = 1024 x 16B chunks, 4 per thread
#pragma unroll
    for (int i = 0; i < 4; ++i) {
      int c = tid + i * 256;
      int row = c >> 3, col = (c & 7) * 8;
      async_cp16(&A[(size_t)(m0 + row) * K + k0 + col], &sA[buf][row][col]);
    }
    // W tile: 128 n-rows x 64 k
#pragma unroll
    for (int i = 0; i < 4; ++i) {
      int c = tid + i * 256;
      int row = c >> 3, col = (c & 7) * 8;
      async_cp16(&WT[(size_t)(n0 + row) * K + k0 + col], &sB[buf][row][col]);
    }
  };

  v8f acc[2][4];
#pragma unroll
  for (int a_ = 0; a_ < 2; ++a_)
#pragma unroll
    for (int b_ = 0; b_ < 4; ++b_)
#pragma unroll
      for (int j = 0; j < 8; ++j) acc[a_][b_][j] = 0.0f;

  stage(0, 0);
  wait_async0();
  __syncthreads();

  const int NKT = K / GBK;
  for (int t = 0; t < NKT; ++t) {
    const int cur = t & 1;
    if (t + 1 < NKT) stage(1 - cur, (t + 1) * GBK);

#pragma unroll
    for (int ks = 0; ks < GBK; ks += 32) {
      // A fragment: lane<16 holds K 0..15, lane>=16 K 16..31
      v16bf af[2];
#pragma unroll
      for (int tm = 0; tm < 2; ++tm) {
        int row = wm * 32 + tm * 16 + r;
#pragma unroll
        for (int j = 0; j < 16; ++j)
          af[tm][j] = sA[cur][row][ks + j + half * 16];
      }
      // B fragment: lanes 0-15 hold K 0..15 (N=lane), lanes 16-31 K 16..31
      v16bf bf_[4];
#pragma unroll
      for (int tn = 0; tn < 4; ++tn) {
        int col = wn * 64 + tn * 16 + r;
#pragma unroll
        for (int j = 0; j < 16; ++j)
          bf_[tn][j] = sB[cur][col][ks + j + half * 16];
      }
#pragma unroll
      for (int tm = 0; tm < 2; ++tm)
#pragma unroll
        for (int tn = 0; tn < 4; ++tn)
          acc[tm][tn] = wmma_bf16(af[tm], bf_[tn], acc[tm][tn]);
    }

    wait_async0();
    __syncthreads();
  }

  // epilogue: VGPR i -> row (i + 8*half), lane r -> col
#pragma unroll
  for (int tm = 0; tm < 2; ++tm)
#pragma unroll
    for (int tn = 0; tn < 4; ++tn)
#pragma unroll
      for (int i = 0; i < 8; ++i) {
        int row = m0 + wm * 32 + tm * 16 + i + 8 * half;
        int col = n0 + wn * 64 + tn * 16 + r;
        float v = acc[tm][tn][i] + bias[col];
        if constexpr (TRANSOUT) {
          int bb = row / Sc, ss = row % Sc;
          int hh = col / Dc, dd = col % Dc;
          C[((size_t)(bb * Hc + hh) * Dc + dd) * Sc + ss] = (TOUT)v;
        } else {
          C[(size_t)row * N + col] = (TOUT)v;
        }
      }
}

// ---------------------------------------------------------------------------
// Flash attention: grid (S/64, H, B), 128 thr = 4 waves, 16 query rows/wave.
// Key tiles of 64, double-buffered async staging.
// Q,K bf16 [B*S, E]; Vt bf16 [(b*H+h)*D + d][s]; O bf16 [B*S, E].
// Softmax scale pre-folded into Q fragments.
// ---------------------------------------------------------------------------
constexpr int QB = 64, TKc = 64;

__global__ __launch_bounds__(128)
void attn_kernel(const __bf16* __restrict__ Q, const __bf16* __restrict__ K,
                 const __bf16* __restrict__ Vt, __bf16* __restrict__ O) {
  __shared__ __bf16 sK[2][TKc][Dc + 8];   // [key][d]
  __shared__ __bf16 sV[2][Dc][TKc + 8];   // [d][key]  (transposed)
  __shared__ __bf16 sP[4][16][TKc + 8];   // per-wave P tile

  const int tid  = threadIdx.x;
  const int wave = tid >> 5;
  const int lane = tid & 31;
  const int half = lane >> 4;
  const int r    = lane & 15;

  const int b  = blockIdx.z;
  const int h  = blockIdx.y;
  const int q0 = blockIdx.x * QB + wave * 16;
  const size_t rowbase = (size_t)b * Sc;
  const size_t vbase   = (size_t)(b * Hc + h) * Dc * Sc;

  auto stageKV = [&](int buf, int kt) {
#pragma unroll
    for (int i = 0; i < 4; ++i) {
      int c = tid + i * 128;
      int row = c >> 3, col = (c & 7) * 8;
      async_cp16(&K[(rowbase + kt + row) * Ec + h * Dc + col], &sK[buf][row][col]);
      async_cp16(&Vt[vbase + (size_t)row * Sc + kt + col],     &sV[buf][row][col]);
    }
  };

  // Q fragments for D=64 (two K-steps), softmax scale folded in
  v16bf qf[2];
#pragma unroll
  for (int ks = 0; ks < 2; ++ks)
#pragma unroll
    for (int j = 0; j < 16; ++j) {
      int d = ks * 32 + j + half * 16;
      qf[ks][j] = (__bf16)((float)Q[(rowbase + q0 + r) * Ec + h * Dc + d] * 0.125f);
    }

  float m_i[8], l_i[8];
#pragma unroll
  for (int i = 0; i < 8; ++i) { m_i[i] = -__builtin_inff(); l_i[i] = 0.0f; }

  v8f oacc[4];
#pragma unroll
  for (int t4 = 0; t4 < 4; ++t4)
#pragma unroll
    for (int j = 0; j < 8; ++j) oacc[t4][j] = 0.0f;

  stageKV(0, 0);
  wait_async0();
  __syncthreads();

  constexpr int NT = Sc / TKc;  // 32
  for (int t = 0; t < NT; ++t) {
    const int cur = t & 1;
    if (t + 1 < NT) stageKV(1 - cur, (t + 1) * TKc);

    // logits tile 16 x 64 = 4 n-tiles, K-dim D=64 in 2 steps
    v8f s_acc[4];
#pragma unroll
    for (int nt = 0; nt < 4; ++nt) {
#pragma unroll
      for (int j = 0; j < 8; ++j) s_acc[nt][j] = 0.0f;
#pragma unroll
      for (int ks = 0; ks < 2; ++ks) {
        v16bf kb;  // B fragment: n = key (nt*16+r), contiguous d per lane
#pragma unroll
        for (int j = 0; j < 16; ++j)
          kb[j] = sK[cur][nt * 16 + r][ks * 32 + j + half * 16];
        s_acc[nt] = wmma_bf16(qf[ks], kb, s_acc[nt]);
      }
    }

    // online softmax; row (i + 8*half) lives in a 16-lane half-group
#pragma unroll
    for (int i = 0; i < 8; ++i) {
      float mx = fmaxf(fmaxf(s_acc[0][i], s_acc[1][i]),
                       fmaxf(s_acc[2][i], s_acc[3][i]));
#pragma unroll
      for (int off = 1; off < 16; off <<= 1)
        mx = fmaxf(mx, __shfl_xor(mx, off, 16));
      float mnew = fmaxf(m_i[i], mx);
      float rs = 0.0f;
#pragma unroll
      for (int nt = 0; nt < 4; ++nt) {
        float p = __expf(s_acc[nt][i] - mnew);
        s_acc[nt][i] = p;
        rs += p;
      }
#pragma unroll
      for (int off = 1; off < 16; off <<= 1)
        rs += __shfl_xor(rs, off, 16);
      float corr = __expf(m_i[i] - mnew);
      l_i[i] = l_i[i] * corr + rs;
      m_i[i] = mnew;
#pragma unroll
      for (int t4 = 0; t4 < 4; ++t4) oacc[t4][i] *= corr;
    }

    // P (C-layout fp32) -> LDS bf16, re-gathered as A fragments
#pragma unroll
    for (int nt = 0; nt < 4; ++nt)
#pragma unroll
      for (int i = 0; i < 8; ++i)
        sP[wave][i + 8 * half][nt * 16 + r] = (__bf16)s_acc[nt][i];
    __syncthreads();

    // O(16x64) += P(16x64) @ V(64x64)
    v16bf pf[2];
#pragma unroll
    for (int ks = 0; ks < 2; ++ks)
#pragma unroll
      for (int j = 0; j < 16; ++j)
        pf[ks][j] = sP[wave][r][ks * 32 + j + half * 16];
#pragma unroll
    for (int nt = 0; nt < 4; ++nt)
#pragma unroll
      for (int ks = 0; ks < 2; ++ks) {
        v16bf vb;  // B fragment: n = d (nt*16+r), contiguous keys per lane
#pragma unroll
        for (int j = 0; j < 16; ++j)
          vb[j] = sV[cur][nt * 16 + r][ks * 32 + j + half * 16];
        oacc[nt] = wmma_bf16(pf[ks], vb, oacc[nt]);
      }

    wait_async0();
    __syncthreads();
  }

  // normalize and store O
#pragma unroll
  for (int nt = 0; nt < 4; ++nt)
#pragma unroll
    for (int i = 0; i < 8; ++i) {
      int row = q0 + i + 8 * half;
      int d   = nt * 16 + r;
      float v = oacc[nt][i] / l_i[i];
      O[(rowbase + row) * Ec + h * Dc + d] = (__bf16)v;
    }
}

// ---------------------------------------------------------------------------
// Launch
// ---------------------------------------------------------------------------
extern "C" void kernel_launch(void* const* d_in, const int* in_sizes, int n_in,
                              void* d_out, int out_size, void* d_ws, size_t ws_size,
                              hipStream_t stream) {
  const float* q    = (const float*)d_in[0];
  const float* k    = (const float*)d_in[1];
  const float* v    = (const float*)d_in[2];
  const float* wq   = (const float*)d_in[3];
  const float* bq   = (const float*)d_in[4];
  const float* wq_a = (const float*)d_in[5];
  const float* wq_b = (const float*)d_in[6];
  const float* wk   = (const float*)d_in[7];
  const float* bk   = (const float*)d_in[8];
  const float* wv   = (const float*)d_in[9];
  const float* bv   = (const float*)d_in[10];
  const float* wv_a = (const float*)d_in[11];
  const float* wv_b = (const float*)d_in[12];
  const float* wo   = (const float*)d_in[13];
  const float* bo   = (const float*)d_in[14];
  float* out = (float*)d_out;

  // workspace layout (64 MB)
  char* ws = (char*)d_ws;
  const size_t MB = 1024 * 1024;
  __bf16* q_bf = (__bf16*)(ws + 0 * MB);    // 8 MB
  __bf16* k_bf = (__bf16*)(ws + 8 * MB);    // 8 MB
  __bf16* v_bf = (__bf16*)(ws + 16 * MB);   // 8 MB
  __bf16* wqT  = (__bf16*)(ws + 24 * MB);   // 2 MB
  __bf16* wkT  = (__bf16*)(ws + 26 * MB);   // 2 MB
  __bf16* wvT  = (__bf16*)(ws + 28 * MB);   // 2 MB
  __bf16* woT  = (__bf16*)(ws + 30 * MB);   // 2 MB
  __bf16* Qp   = (__bf16*)(ws + 32 * MB);   // 8 MB
  __bf16* Kp   = (__bf16*)(ws + 40 * MB);   // 8 MB
  __bf16* VpT  = (__bf16*)(ws + 48 * MB);   // 8 MB
  __bf16* Og   = (__bf16*)(ws + 56 * MB);   // 8 MB

  // 1) convert activations; transpose/convert weights (LoRA folded for Q/V)
  {
    int n = Mc * Ec;
    dim3 g(n / 4 / 256);
    cvt_bf16_kernel<<<g, 256, 0, stream>>>(q, q_bf, n);
    cvt_bf16_kernel<<<g, 256, 0, stream>>>(k, k_bf, n);
    cvt_bf16_kernel<<<g, 256, 0, stream>>>(v, v_bf, n);
  }
  {
    dim3 gt(Ec / 32, Ec / 32);
    wtrans_kernel<<<gt, 256, 0, stream>>>(wk, wkT);
    wtrans_kernel<<<gt, 256, 0, stream>>>(wo, woT);
    dim3 gf((Ec * Ec + 255) / 256);
    lora_fold_t_kernel<<<gf, 256, 0, stream>>>(wq, wq_a, wq_b, wqT);
    lora_fold_t_kernel<<<gf, 256, 0, stream>>>(wv, wv_a, wv_b, wvT);
  }

  // 2) projections
  {
    dim3 g(Ec / GBN, Mc / GBM);
    gemm_bias_kernel<__bf16, false><<<g, 256, 0, stream>>>(q_bf, wqT, bq, Qp, Mc, Ec, Ec);
    gemm_bias_kernel<__bf16, false><<<g, 256, 0, stream>>>(k_bf, wkT, bk, Kp, Mc, Ec, Ec);
    gemm_bias_kernel<__bf16, true ><<<g, 256, 0, stream>>>(v_bf, wvT, bv, VpT, Mc, Ec, Ec);
  }

  // 3) attention
  {
    dim3 g(Sc / QB, Hc, Bc);
    attn_kernel<<<g, 128, 0, stream>>>(Qp, Kp, VpT, Og);
  }

  // 4) output projection (fp32 out)
  {
    dim3 g(Ec / GBN, Mc / GBM);
    gemm_bias_kernel<float, false><<<g, 256, 0, stream>>>(Og, woT, bo, out, Mc, Ec, Ec);
  }
}
